// MyConv2D_67843303407704
// MI455X (gfx1250) — matrix operand order repaired and checked by reading, not compiled, not run on hardware
//
#include <hip/hip_runtime.h>
#include <hip/hip_bf16.h>

// Implicit-GEMM 3x3 conv for gfx1250 (MI455X), bf16x3 split-precision fp32
// emulation on V_WMMA_F32_16X16X32_BF16, with Tensor-Data-Mover (TDM) staging
// of the pre-split weight tiles into LDS (tensor_load_to_lds + TENSORcnt).
//   x = x_hi + x_lo (bf16);  x*w ~= hi*hi + hi*lo + lo*hi  (f32 accumulate)
// GEMM view: C[M=C_out=256][N=pixels=100352], K = C_in*3*3 = 1152.
// Block: 256 threads (8 wave32). Block tile: 128 C_out x 64 pixels.
// K staged through LDS in double-buffered chunks of 64.

typedef __attribute__((ext_vector_type(8)))  float        v8f;
typedef __attribute__((ext_vector_type(8)))  __bf16       v8bf;
typedef __attribute__((ext_vector_type(16))) __bf16       v16bf;
typedef __attribute__((ext_vector_type(4)))  unsigned int v4u;
typedef __attribute__((ext_vector_type(8)))  int          v8i;
typedef __attribute__((ext_vector_type(4)))  int          v4i;
typedef unsigned short ushort_t;

#define IN_C   128
#define OUT_C  256
#define HW     56
#define HW2    (HW * HW)        // 3136
#define NPIX   (32 * HW2)       // 100352
#define KDIM   (IN_C * 9)       // 1152
#define KC     64               // K chunk per LDS stage
#define NCHUNK (KDIM / KC)      // 18
#define PIXT   64               // pixel tile per block
#define COUTT  128              // cout tile per block
#define PADK   72               // padded row stride in bf16 elems (144B, 16B-aligned)

// Round-to-nearest-even fp32 -> bf16 split into hi + lo halves (bit ops only).
__device__ __forceinline__ void bf16_split(float f, unsigned short& h, unsigned short& l) {
    unsigned int u  = __float_as_uint(f);
    unsigned int hr = (u + 0x7FFFu + ((u >> 16) & 1u)) & 0xFFFF0000u;
    h = (unsigned short)(hr >> 16);
    float flo = f - __uint_as_float(hr);
    unsigned int ul = __float_as_uint(flo);
    l = (unsigned short)((ul + 0x7FFFu + ((ul >> 16) & 1u)) >> 16);
}

// A fragment (16-bit 16x32): per lane, runs [ks+hh*8 .. +7] and [ks+16+hh*8 .. +7].
__device__ __forceinline__ v16bf fragA(const unsigned short* p) {
    v8bf r0 = *(const v8bf*)(p);
    v8bf r1 = *(const v8bf*)(p + 16);
    return __builtin_shufflevector(r0, r1, 0,1,2,3,4,5,6,7,8,9,10,11,12,13,14,15);
}
// B fragment (16-bit 32x16): per lane, runs [ks+hh*8 .. +7] and [ks+16+hh*8 .. +7].
__device__ __forceinline__ v16bf fragB(const unsigned short* p) {
    v8bf r0 = *(const v8bf*)(p);
    v8bf r1 = *(const v8bf*)(p + 16);
    return __builtin_shufflevector(r0, r1, 0,1,2,3,4,5,6,7,8,9,10,11,12,13,14,15);
}

// TDM: DMA a 2-D bf16 tile (64 wide x 128 rows, row pitch KDIM elems) from
// global memory into LDS, inserting 4 dwords of padding after every 32 dwords
// (one 128B row) so rows land at the PADK=72-element stride.
// D# fields per CDNA5 ISA ch.8 (group0/1; groups 2/3 zero for 2-D tensors).
__device__ __forceinline__ void tdm_load_tile(unsigned int lds_addr, const void* gptr) {
    unsigned long long ga = (unsigned long long)(uintptr_t)gptr;
    v4u g0 = { 1u,                                     // count=1 (valid descriptor)
               lds_addr,                               // lds_addr (bytes)
               (unsigned int)ga,                       // global_addr[31:0]
               (unsigned int)((ga >> 32) & 0x01FFFFFFu) | (2u << 30) }; // ga[56:32] | type=2
    v8i g1 = { (int)((1u << 16)     // data_size = 1 -> 2 bytes
                   | (1u << 20)     // pad_enable
                   | (4u << 22)     // pad_interval: 32 dwords (one row)
                   | (3u << 25)),   // pad_amount: 4 dwords (8 bf16)
               (int)(64u << 16),    // tensor_dim0[15:0] = 64 (in d1[31:16])
               (int)(128u << 16),   // tensor_dim1[15:0] = 128 (in d2[31:16])
               (int)(64u << 16),    // tile_dim0 = 64 (in d3[31:16])
               128,                 // tile_dim1 = 128
               KDIM,                // tensor_dim0_stride[31:0] = 1152
               0, 0 };
    v4i z4 = { 0, 0, 0, 0 };
#if defined(__clang_major__) && (__clang_major__ >= 23)
    v8i z8 = { 0, 0, 0, 0, 0, 0, 0, 0 };
    __builtin_amdgcn_tensor_load_to_lds(g0, g1, z4, z4, z8, 0);
#else
    __builtin_amdgcn_tensor_load_to_lds(g0, g1, z4, z4, 0);
#endif
}

// One-time weight preprocessing: split W (fp32, [256][1152]) into hi/lo bf16.
__global__ __launch_bounds__(256)
void wprep_kernel(const float* __restrict__ w,
                  unsigned short* __restrict__ wh,
                  unsigned short* __restrict__ wl) {
    int idx = blockIdx.x * 256 + threadIdx.x;   // grid sized exactly: 256*1152 elems
    unsigned short h, l;
    bf16_split(w[idx], h, l);
    wh[idx] = h;
    wl[idx] = l;
}

template <bool USE_WS>
__global__ __launch_bounds__(256)
void conv3x3_wmma_bf16x3_kernel(const float* __restrict__ x,
                                const float* __restrict__ w,
                                const unsigned short* __restrict__ wh,
                                const unsigned short* __restrict__ wl,
                                const float* __restrict__ bias,
                                float* __restrict__ out) {
    // Double-buffered hi/lo staging. Total LDS = 110,592 B (<320KB/WGP).
    __shared__ __align__(16) unsigned short sAh[2][COUTT * PADK];
    __shared__ __align__(16) unsigned short sAl[2][COUTT * PADK];
    __shared__ __align__(16) unsigned short sBh[2][PIXT * PADK];
    __shared__ __align__(16) unsigned short sBl[2][PIXT * PADK];

    const int tid    = threadIdx.x;
    const int lane   = tid & 31;
    const int wave   = tid >> 5;       // 0..7
    const int lane16 = lane & 15;
    const int hiHalf = lane >> 4;      // 0 or 1

    const int pixBase = blockIdx.x * PIXT;   // 1568 tiles (exact)
    const int oBase   = blockIdx.y * COUTT;  // 2 tiles (exact)

    const float* Wp = w + (size_t)oBase * KDIM;
    const unsigned short* WhP = wh + (size_t)oBase * KDIM;
    const unsigned short* WlP = wl + (size_t)oBase * KDIM;

    // ---- Hoisted im2col pixel math: each thread gathers one pixel, 16 k's/chunk ----
    const int pB  = tid & 63;          // pixel within tile (fixed per thread)
    const int kB  = tid >> 6;          // k row base (0..3), step 4
    const int qB  = pixBase + pB;
    const int nB  = qB / HW2;
    const int rB  = qB - nB * HW2;
    const int ohB = rB / HW;
    const int owB = rB - ohB * HW;
    const float* xn = x + (size_t)nB * IN_C * HW2;

    const int kqB = tid & 15;          // float4 column for fallback A staging

    v8f acc0 = {}, acc1 = {}, acc2 = {}, acc3 = {};

    // Stage one K-chunk into LDS buffer `buf`.
    auto stage = [&](int buf, int k0) {
        if constexpr (USE_WS) {
            // A tiles via Tensor Data Mover: wave 0 issues two 2-D DMA descriptors
            // (hi and lo). EXEC is ignored by tensor ops; TENSORcnt tracks them.
            if (wave == 0) {
                tdm_load_tile((unsigned int)(uintptr_t)&sAh[buf][0], WhP + k0);
                tdm_load_tile((unsigned int)(uintptr_t)&sAl[buf][0], WlP + k0);
            }
        } else {
            // Fallback: fp32 weights, split in-kernel (coalesced float4 loads).
            #pragma unroll
            for (int j = 0; j < 8; ++j) {
                int row = (tid >> 4) + j * 16;     // 0..127
                float4 v = *(const float4*)(Wp + (size_t)row * KDIM + k0 + kqB * 4);
                unsigned short h0, l0, h1, l1, h2, l2, h3, l3;
                bf16_split(v.x, h0, l0);
                bf16_split(v.y, h1, l1);
                bf16_split(v.z, h2, l2);
                bf16_split(v.w, h3, l3);
                uint2 hp = make_uint2((unsigned)h0 | ((unsigned)h1 << 16),
                                      (unsigned)h2 | ((unsigned)h3 << 16));
                uint2 lp = make_uint2((unsigned)l0 | ((unsigned)l1 << 16),
                                      (unsigned)l2 | ((unsigned)l3 << 16));
                *(uint2*)&sAh[buf][row * PADK + kqB * 4] = hp;
                *(uint2*)&sAl[buf][row * PADK + kqB * 4] = lp;
            }
        }
        // im2col: 64 k x 64 pixels, 16 scalars per thread (pixel fixed per thread).
        #pragma unroll
        for (int j = 0; j < 16; ++j) {
            int k  = kB + j * 4;               // 0..63
            int kg = k0 + k;
            int c   = kg / 9;
            int rem = kg - c * 9;
            int kh  = rem / 3;
            int kw  = rem - kh * 3;
            int ih = ohB + kh - 1;
            int iw = owB + kw - 1;
            float val = 0.0f;
            if ((unsigned)ih < (unsigned)HW && (unsigned)iw < (unsigned)HW) {
                val = xn[((size_t)c * HW + ih) * HW + iw];
            }
            unsigned short h, l;
            bf16_split(val, h, l);
            sBh[buf][pB * PADK + k] = h;
            sBl[buf][pB * PADK + k] = l;
        }
    };

    stage(0, 0);
    if (USE_WS && wave == 0) {
        __builtin_amdgcn_s_wait_tensorcnt(0);  // chunk 0 tiles resident before barrier
    }

    const int aRow = (wave * 16 + lane16) * PADK + hiHalf * 8;
    const int bOff = lane16 * PADK + hiHalf * 8;

    for (int i = 0; i < NCHUNK; ++i) {
        __syncthreads();
        const int cur = i & 1;
        if (i + 1 < NCHUNK) {
            stage(cur ^ 1, (i + 1) * KC);      // DMA + gather overlap the WMMAs below
        }

        const unsigned short* Ah = &sAh[cur][aRow];
        const unsigned short* Al = &sAl[cur][aRow];
        const unsigned short* Bh = &sBh[cur][bOff];
        const unsigned short* Bl = &sBl[cur][bOff];

        #pragma unroll
        for (int ks = 0; ks < KC; ks += 32) {
            v16bf ah = fragA(Ah + ks);
            v16bf al = fragA(Al + ks);
            v16bf bh0 = fragB(Bh + ks);
            v16bf bh1 = fragB(Bh + ks + 16 * PADK);
            v16bf bh2 = fragB(Bh + ks + 32 * PADK);
            v16bf bh3 = fragB(Bh + ks + 48 * PADK);
            v16bf bl0 = fragB(Bl + ks);
            v16bf bl1 = fragB(Bl + ks + 16 * PADK);
            v16bf bl2 = fragB(Bl + ks + 32 * PADK);
            v16bf bl3 = fragB(Bl + ks + 48 * PADK);
            // Terms interleaved across 4 independent accumulator chains.
            acc0 = __builtin_amdgcn_wmma_f32_16x16x32_bf16(false, ah, false, bh0, (short)0, acc0, false, false);
            acc1 = __builtin_amdgcn_wmma_f32_16x16x32_bf16(false, ah, false, bh1, (short)0, acc1, false, false);
            acc2 = __builtin_amdgcn_wmma_f32_16x16x32_bf16(false, ah, false, bh2, (short)0, acc2, false, false);
            acc3 = __builtin_amdgcn_wmma_f32_16x16x32_bf16(false, ah, false, bh3, (short)0, acc3, false, false);
            acc0 = __builtin_amdgcn_wmma_f32_16x16x32_bf16(false, ah, false, bl0, (short)0, acc0, false, false);
            acc1 = __builtin_amdgcn_wmma_f32_16x16x32_bf16(false, ah, false, bl1, (short)0, acc1, false, false);
            acc2 = __builtin_amdgcn_wmma_f32_16x16x32_bf16(false, ah, false, bl2, (short)0, acc2, false, false);
            acc3 = __builtin_amdgcn_wmma_f32_16x16x32_bf16(false, ah, false, bl3, (short)0, acc3, false, false);
            acc0 = __builtin_amdgcn_wmma_f32_16x16x32_bf16(false, al, false, bh0, (short)0, acc0, false, false);
            acc1 = __builtin_amdgcn_wmma_f32_16x16x32_bf16(false, al, false, bh1, (short)0, acc1, false, false);
            acc2 = __builtin_amdgcn_wmma_f32_16x16x32_bf16(false, al, false, bh2, (short)0, acc2, false, false);
            acc3 = __builtin_amdgcn_wmma_f32_16x16x32_bf16(false, al, false, bh3, (short)0, acc3, false, false);
        }

        if (USE_WS && wave == 0) {
            // Tiles DMA'd this iteration must be resident before next barrier.
            __builtin_amdgcn_s_wait_tensorcnt(0);
        }
    }

    // ---- Epilogue: D layout: VGPR v -> M=v (lanes 0-15) / M=v+8 (lanes 16-31),
    //      N = lane%16. Add bias, store NCHW (64B-contiguous per 16-lane group). ----
    const int mOff = hiHalf * 8;
    v8f accs[4] = {acc0, acc1, acc2, acc3};
    #pragma unroll
    for (int g = 0; g < 4; ++g) {
        int q  = pixBase + g * 16 + lane16;
        int n  = q / HW2;
        int r  = q - n * HW2;
        size_t outPix = (size_t)n * OUT_C * HW2 + r;
        #pragma unroll
        for (int v = 0; v < 8; ++v) {
            int co = oBase + wave * 16 + v + mOff;
            out[outPix + (size_t)co * HW2] = accs[g][v] + bias[co];
        }
    }
}

extern "C" void kernel_launch(void* const* d_in, const int* in_sizes, int n_in,
                              void* d_out, int out_size, void* d_ws, size_t ws_size,
                              hipStream_t stream) {
    const float* x    = (const float*)d_in[0];
    const float* W    = (const float*)d_in[1];
    const float* bias = (const float*)d_in[2];
    float* out = (float*)d_out;

    dim3 grid(NPIX / PIXT, OUT_C / COUTT);  // 1568 x 2
    dim3 block(256);

    const size_t wsNeed = (size_t)OUT_C * KDIM * 2 * sizeof(unsigned short); // 1.13 MB
    if (ws_size >= wsNeed) {
        unsigned short* wh = (unsigned short*)d_ws;
        unsigned short* wl = wh + (size_t)OUT_C * KDIM;
        wprep_kernel<<<dim3((OUT_C * KDIM) / 256), block, 0, stream>>>(W, wh, wl);
        conv3x3_wmma_bf16x3_kernel<true><<<grid, block, 0, stream>>>(x, W, wh, wl, bias, out);
    } else {
        conv3x3_wmma_bf16x3_kernel<false><<<grid, block, 0, stream>>>(x, W, nullptr, nullptr, bias, out);
    }
}
